// NuADMInference_3685081940030
// MI455X (gfx1250) — matrix operand order repaired and checked by reading, compiled, not run on hardware
//
#include <hip/hip_runtime.h>
#include <math.h>

// ---------------- problem constants (from reference) ----------------
#define NRES  2048
#define KNB   32
#define DIM   256
#define PD    128
#define HEADS 8
#define DH    32
#define NBINS 32
#define RBFB  64
#define NK    (NRES*KNB)

typedef __attribute__((ext_vector_type(16))) __bf16 v16bf;
typedef __attribute__((ext_vector_type(8)))  __bf16 v8bf;
typedef __attribute__((ext_vector_type(8)))  float  v8f;

__device__ __forceinline__ float geluf(float x) {
    return 0.5f * x * (1.0f + tanhf(0.7978845608028654f * (x + 0.044715f * x * x * x)));
}

// ---------------------------------------------------------------------------
// Generic bf16 WMMA GEMM:  out = epilogue(A[M,K] @ B[K,N])
// block = 128 threads (4 waves); wave w -> rows m0+16w, cols n0..n0+63.
// B K-slabs (32x64 bf16 = 4KB) are staged global->LDS with the CDNA5 async
// copy path (GLOBAL_LOAD_ASYNC_TO_LDS_B128, ASYNCcnt), double-buffered, and
// shared by all 4 waves; fragments are then read as single bf16 elements.
// Requires M%64==0, N%64==0, K%32==0 (true for all GEMMs here).
// Epilogue: +bias[n], gelu, row-mask (zero acc), +residual, store f32/bf16.
// ---------------------------------------------------------------------------
__global__ void __launch_bounds__(128)
wmma_gemm(const __bf16* __restrict__ A, const __bf16* __restrict__ B,
          float* outF, __bf16* outB,
          const float* bias, const float* residual,
          const unsigned char* rowmask, int rm_div,
          int Mm, int Nn, int Kk, int do_gelu)
{
    __shared__ __align__(32) __bf16 Bs[2][32 * 64];   // two 4KB K-slabs

    int tid  = threadIdx.x;
    int wave = tid >> 5;
    int lane = tid & 31;
    int m0 = blockIdx.y * 64 + wave * 16;
    int n0 = blockIdx.x * 64;
    int arow  = lane & 15;
    int kbase = (lane >> 4) << 3;           // 0 or 8

    v8f acc[4];
#pragma unroll
    for (int t = 0; t < 4; ++t)
#pragma unroll
        for (int e = 0; e < 8; ++e) acc[t][e] = 0.0f;

    // cooperative async stage of B[k0..k0+31, n0..n0+63] into Bs[buf]
    // 128 threads x 32B each: thread -> row (tid>>2), 32B chunk (tid&3)
    auto stage = [&](int buf, int k0) {
        int r  = tid >> 2;
        int ce = (tid & 3) * 16;                       // element offset in row
        const __bf16* g = B + (size_t)(k0 + r) * Nn + n0 + ce;
        unsigned l = (unsigned)(uintptr_t)(&Bs[buf][0]) + (unsigned)(r * 128 + (tid & 3) * 32);
        unsigned long long ga = (unsigned long long)(uintptr_t)g;
        asm volatile("global_load_async_to_lds_b128 %0, %1, off"
                     :: "v"(l), "v"(ga) : "memory");
        asm volatile("global_load_async_to_lds_b128 %0, %1, off offset:16"
                     :: "v"(l), "v"(ga) : "memory");
    };

    int nslab = Kk >> 5;
    stage(0, 0);
    for (int s = 0; s < nslab; ++s) {
        int k0 = s << 5;
        asm volatile("s_wait_asynccnt 0" ::: "memory"); // own async copies done
        __syncthreads();                                 // slab visible to all waves
        if (s + 1 < nslab) stage((s + 1) & 1, k0 + 32);  // prefetch next slab

        // A fragment: lane holds row (m0+arow), K = {kbase..+7, kbase+16..+23}
        const __bf16* Ap = A + (size_t)(m0 + arow) * Kk + k0 + kbase;
        v8bf alo = *(const v8bf*)(Ap);
        v8bf ahi = *(const v8bf*)(Ap + 16);
        v16bf av;
#pragma unroll
        for (int e = 0; e < 8; ++e) { av[e] = alo[e]; av[8 + e] = ahi[e]; }

        // B fragments from LDS: lane holds column (lane&15), K = {kbase..+7, kbase+16..+23}
        const __bf16* Bl = &Bs[s & 1][kbase * 64 + (lane & 15)];
#pragma unroll
        for (int t = 0; t < 4; ++t) {
            v16bf bv;
#pragma unroll
            for (int e = 0; e < 8; ++e) { bv[e] = Bl[e * 64 + t * 16]; bv[8 + e] = Bl[(16 + e) * 64 + t * 16]; }
            acc[t] = __builtin_amdgcn_wmma_f32_16x16x32_bf16(
                false, av, false, bv, (short)0, acc[t], false, false);
        }
    }

    // C/D layout: VGPR j -> row j (+8 for lanes>=16), col = lane&15
    int crow = (lane >> 4) << 3;
    int ccol = lane & 15;
#pragma unroll
    for (int t = 0; t < 4; ++t) {
#pragma unroll
        for (int jr = 0; jr < 8; ++jr) {
            int m = m0 + crow + jr;
            int nn = n0 + t * 16 + ccol;
            float v = acc[t][jr];
            if (bias)    v += bias[nn];
            if (do_gelu) v = geluf(v);
            if (rowmask && rowmask[m / rm_div] == 0) v = 0.0f;
            size_t o = (size_t)m * Nn + nn;
            if (residual) v += residual[o];
            if (outF) outF[o] = v;
            if (outB) outB[o] = (__bf16)v;
        }
    }
}

// ---------------- f32 -> bf16 convert ----------------
__global__ void f2bf_kernel(const float* __restrict__ in, __bf16* __restrict__ out, int n)
{
    int i = blockIdx.x * 256 + threadIdx.x;
    if (i < n) out[i] = (__bf16)in[i];
}

// ---------------- LayerNorm (one block per row, blockDim == Dd) ----------------
__global__ void ln_kernel(const float* __restrict__ in, const float* __restrict__ g,
                          const float* __restrict__ b, __bf16* __restrict__ out, int Dd)
{
    extern __shared__ float sm[];
    size_t row = blockIdx.x;
    int t = threadIdx.x;
    float x = in[row * Dd + t];
    sm[t] = x;
    __syncthreads();
    for (int s = Dd >> 1; s > 0; s >>= 1) { if (t < s) sm[t] += sm[t + s]; __syncthreads(); }
    float mean = sm[0] / (float)Dd;
    __syncthreads();
    float dx = x - mean;
    sm[t] = dx * dx;
    __syncthreads();
    for (int s = Dd >> 1; s > 0; s >>= 1) { if (t < s) sm[t] += sm[t + s]; __syncthreads(); }
    float var = sm[0] / (float)Dd;
    out[row * Dd + t] = (__bf16)(dx * rsqrtf(var + 1e-5f) * g[t] + b[t]);
}

// ---------------- pair = rel@w_rel (one-hot lookup) + fi + fj[nb] ----------------
__global__ void pair_build(const int* __restrict__ nb, const int* __restrict__ resi,
                           const int* __restrict__ chain, const int* __restrict__ batch,
                           const float* __restrict__ w_rel,
                           const float* __restrict__ fi, const float* __restrict__ fj,
                           float* __restrict__ pair)
{
    int idx = blockIdx.x;             // n*KNB + k
    int n = idx >> 5;
    int c = threadIdx.x;              // 0..127
    int j = nb[idx];
    int ja = j < 0 ? j + NRES : j;    // JAX negative-index wrap
    int d = resi[ja] - resi[n];
    d = d < -32 ? -32 : (d > 32 ? 32 : d);
    d += 32;
    bool same = (chain[ja] == chain[n]) && (batch[ja] == batch[n]);
    float v = same ? w_rel[d * PD + c] : 0.0f;
    v += fi[(size_t)n * PD + c] + fj[(size_t)ja * PD + c];
    pair[(size_t)idx * PD + c] = v;
}

// ---------------- neighbour attention (shared by both attention blocks) --------
// bias = pair @ wb ; pm gating; softmax over K; o = sum_k att*(v[nb]+pvv)
__global__ void attn_kernel(const float* __restrict__ q, const float* __restrict__ kmat,
                            const float* __restrict__ vmat, const int* __restrict__ nb,
                            const __bf16* __restrict__ pairb, const float* __restrict__ wb,
                            const __bf16* __restrict__ pvv,
                            const unsigned char* __restrict__ mask,
                            const unsigned char* __restrict__ posmask,
                            __bf16* __restrict__ out)
{
    __shared__ float s_att[HEADS * KNB];
    __shared__ float s_red[HEADS];
    __shared__ int   s_ja[KNB];
    __shared__ float s_pm[KNB];

    int n = blockIdx.x;
    int t = threadIdx.x;              // 256 threads
    int kk = t >> 3;                  // neighbour 0..31
    int h  = t & 7;                   // head 0..7
    int j = nb[n * KNB + kk];
    int ja = j < 0 ? j + NRES : j;
    if (h == 0) {
        s_ja[kk] = ja;
        float pm = (mask[n] && mask[ja] && (j != -1)) ? 1.0f : 0.0f;
        if (posmask && posmask[n] == 0) pm = 0.0f;
        s_pm[kk] = pm;
    }
    // pair bias
    const __bf16* pr = pairb + (size_t)(n * KNB + kk) * PD;
    __builtin_prefetch(pr, 0, 0);                    // global_prefetch_b8
    float bias = 0.0f;
    for (int p = 0; p < PD; ++p) bias += (float)pr[p] * wb[p * HEADS + h];
    // q.k
    const float* qp = q    + (size_t)n  * DIM + h * DH;
    const float* kp = kmat + (size_t)ja * DIM + h * DH;
    float dot = 0.0f;
#pragma unroll
    for (int d = 0; d < DH; ++d) dot += qp[d] * kp[d];
    float logit = dot * 0.17677669529663687f + bias;   // 1/sqrt(32)
    __syncthreads();
    if (s_pm[kk] <= 0.0f) logit = -1e9f;
    s_att[h * KNB + kk] = logit;
    __syncthreads();
    if (t < HEADS) {
        float mx = -3.4e38f;
        for (int i = 0; i < KNB; ++i) mx = fmaxf(mx, s_att[t * KNB + i]);
        s_red[t] = mx;
    }
    __syncthreads();
    float e = __expf(s_att[h * KNB + kk] - s_red[h]);
    s_att[h * KNB + kk] = e;
    __syncthreads();
    if (t < HEADS) {
        float sum = 0.0f;
        for (int i = 0; i < KNB; ++i) sum += s_att[t * KNB + i];
        s_red[t] = sum;
    }
    __syncthreads();
    // output: t -> (h2, d)
    int h2 = t >> 5, d = t & 31;
    float inv = 1.0f / s_red[h2];
    float accv = 0.0f;
    for (int k2 = 0; k2 < KNB; ++k2) {
        float a = s_att[h2 * KNB + k2] * inv;
        int jj = s_ja[k2];
        float vv = vmat[(size_t)jj * DIM + h2 * DH + d];
        if (pvv) vv += (float)pvv[(size_t)(n * KNB + k2) * DIM + h2 * DH + d];
        accv += a * vv;
    }
    out[(size_t)n * DIM + t] = (__bf16)accv;
}

// ---------------- rigid frames from N/CA/C ----------------
__global__ void frames_kernel(const float* __restrict__ pos, float* __restrict__ R,
                              float* __restrict__ ca)
{
    int n = blockIdx.x * 64 + threadIdx.x;
    if (n >= NRES) return;
    const float* p0 = pos + (size_t)n * 14 * 3;
    float Nx=p0[0],Ny=p0[1],Nz=p0[2];
    float Ax=p0[3],Ay=p0[4],Az=p0[5];
    float Cx=p0[6],Cy=p0[7],Cz=p0[8];
    float e1x=Cx-Ax,e1y=Cy-Ay,e1z=Cz-Az;
    float n1 = sqrtf(e1x*e1x+e1y*e1y+e1z*e1z) + 1e-6f;
    e1x/=n1; e1y/=n1; e1z/=n1;
    float ux=Nx-Ax,uy=Ny-Ay,uz=Nz-Az;
    float dp = ux*e1x+uy*e1y+uz*e1z;
    float e2x=ux-dp*e1x, e2y=uy-dp*e1y, e2z=uz-dp*e1z;
    float n2 = sqrtf(e2x*e2x+e2y*e2y+e2z*e2z) + 1e-6f;
    e2x/=n2; e2y/=n2; e2z/=n2;
    float e3x=e1y*e2z-e1z*e2y, e3y=e1z*e2x-e1x*e2z, e3z=e1x*e2y-e1y*e2x;
    ca[n*3+0]=Ax; ca[n*3+1]=Ay; ca[n*3+2]=Az;
    float* r = R + (size_t)n * 9;        // R[n][j][a] = e_a[j]
    r[0]=e1x; r[1]=e2x; r[2]=e3x;
    r[3]=e1y; r[4]=e2y; r[5]=e3y;
    r[6]=e1z; r[7]=e2z; r[8]=e3z;
}

// ---------------- geometric pair features (ppair pre-LN) ----------------
__global__ void ppair_build(const float* __restrict__ pos, const float* __restrict__ logits,
                            const float* __restrict__ R, const float* __restrict__ ca,
                            const int* __restrict__ nb, const int* __restrict__ resi,
                            const int* __restrict__ chain, const int* __restrict__ batch,
                            const unsigned char* __restrict__ mask,
                            const unsigned char* __restrict__ posmask,
                            const float* __restrict__ w_rel2, const float* __restrict__ w_log,
                            const float* __restrict__ w_rbf, const float* __restrict__ w_dir,
                            const float* __restrict__ w_rot, const float* __restrict__ w_vec,
                            float* __restrict__ outp)
{
    __shared__ float sl[NBINS], srbf[RBFB], sdir[3], srot[9], spv[9];
    __shared__ float sdvec[3], sdist, sred[2];

    int idx = blockIdx.x;
    int n = idx >> 5;
    int t = threadIdx.x;              // 128 threads
    int j = nb[idx];
    int ja = j < 0 ? j + NRES : j;

    if (t == 0) {
        float dx = ca[ja*3+0]-ca[n*3+0];
        float dy = ca[ja*3+1]-ca[n*3+1];
        float dz = ca[ja*3+2]-ca[n*3+2];
        sdvec[0]=dx; sdvec[1]=dy; sdvec[2]=dz;
        sdist = sqrtf(dx*dx+dy*dy+dz*dz);
    }
    __syncthreads();
    float dist = sdist;
    const float* r1 = R + (size_t)n  * 9;
    const float* r2 = R + (size_t)ja * 9;
    if (t < RBFB) {
        float step = 22.0f / (float)RBFB;
        float c = step * t + step * 0.5f;
        float z = (dist - c) / step;
        srbf[t] = __expf(-z * z);
    } else if (t < 67) {
        int a = t - 64;
        float inv = 1.0f / (dist + 1e-6f);
        sdir[a] = (r1[0*3+a]*sdvec[0] + r1[1*3+a]*sdvec[1] + r1[2*3+a]*sdvec[2]) * inv;
    } else if (t < 76) {
        int ab = t - 67; int a = ab / 3, b = ab % 3;
        srot[ab] = r1[0*3+a]*r2[0*3+b] + r1[1*3+a]*r2[1*3+b] + r1[2*3+a]*r2[2*3+b];
    } else if (t < 85) {
        int ma = t - 76; int m = ma / 3, a = ma % 3;
        float px = pos[((size_t)ja*14 + m)*3+0] - ca[n*3+0];
        float py = pos[((size_t)ja*14 + m)*3+1] - ca[n*3+1];
        float pz = pos[((size_t)ja*14 + m)*3+2] - ca[n*3+2];
        spv[ma] = r1[0*3+a]*px + r1[1*3+a]*py + r1[2*3+a]*pz;
    } else if (t >= 96) {
        int b = t - 96;                       // 0..31
        sl[b] = logits[((size_t)n * NRES + ja) * NBINS + b];
    }
    __syncthreads();
    if (t == 0) { float mx=-3.4e38f; for (int i=0;i<NBINS;++i) mx=fmaxf(mx,sl[i]); sred[0]=mx; }
    __syncthreads();
    if (t < NBINS) sl[t] = __expf(sl[t] - sred[0]);
    __syncthreads();
    if (t == 0) { float s=0.0f; for (int i=0;i<NBINS;++i) s+=sl[i]; sred[1]=s; }
    __syncthreads();

    float pm2 = (mask[n] && mask[ja] && (j != -1) && posmask[n]) ? 1.0f : 0.0f;
    float slscale = pm2 / sred[1];
    int d = resi[ja] - resi[n];
    d = d < -32 ? -32 : (d > 32 ? 32 : d);
    d += 32;
    bool same = (chain[ja] == chain[n]) && (batch[ja] == batch[n]);

    int c = t;
    float acc = same ? w_rel2[d * PD + c] : 0.0f;
    for (int b = 0; b < NBINS; ++b) acc += sl[b] * slscale * w_log[b * PD + c];
    for (int b = 0; b < RBFB;  ++b) acc += srbf[b] * w_rbf[b * PD + c];
#pragma unroll
    for (int i = 0; i < 3; ++i) acc += sdir[i] * w_dir[i * PD + c];
#pragma unroll
    for (int i = 0; i < 9; ++i) acc += srot[i] * w_rot[i * PD + c];
#pragma unroll
    for (int i = 0; i < 9; ++i) acc += spv[i] * w_vec[i * PD + c];
    outp[(size_t)idx * PD + c] = acc;
}

// ---------------- segment mask counts: cnt[0..7]=chain, cnt[8..11]=batch --------
__global__ void counts_kernel(const int* __restrict__ chain, const int* __restrict__ batch,
                              const unsigned char* __restrict__ mask, float* __restrict__ cnt)
{
    int t = threadIdx.x;
    if (t < 8) {
        float c = 0.0f;
        for (int n = 0; n < NRES; ++n) if (chain[n] == t && mask[n]) c += 1.0f;
        cnt[t] = c;
    } else if (t < 12) {
        int s = t - 8;
        float c = 0.0f;
        for (int n = 0; n < NRES; ++n) if (batch[n] == s && mask[n]) c += 1.0f;
        cnt[8 + s] = c;
    }
}

// ---------------- segment mean of gk (x) gv outer products ----------------
// block: 64 threads (j); blockIdx decodes (seg, head, i)
__global__ void kvseg_kernel(const float* __restrict__ gk, const float* __restrict__ gv,
                             const int* __restrict__ seg, const unsigned char* __restrict__ mask,
                             const float* __restrict__ cnt, int cntoff, float* __restrict__ outm)
{
    int b = blockIdx.x;
    int jj = threadIdx.x;
    int i = b & 63;
    int h = (b >> 6) & 7;
    int s = b >> 9;
    float acc = 0.0f;
    for (int n = 0; n < NRES; ++n) {
        if (seg[n] == s && mask[n])
            acc += gk[(size_t)n * 512 + h * 64 + i] * gv[(size_t)n * 512 + h * 64 + jj];
    }
    outm[((((size_t)s * 8 + h) * 64 + i) * 64) + jj] = acc / fmaxf(cnt[cntoff + s], 1e-6f);
}

// ---------------- batch-mean of features@gwb ----------------
__global__ void gbmean_kernel(const float* __restrict__ gwbo, const int* __restrict__ batch,
                              const unsigned char* __restrict__ mask,
                              const float* __restrict__ cnt, float* __restrict__ gbm)
{
    int s = blockIdx.x;
    int t = threadIdx.x;
    for (int r = 0; r < 4; ++r) {
        int c = t + r * 256;
        float acc = 0.0f;
        for (int n = 0; n < NRES; ++n)
            if (batch[n] == s && mask[n]) acc += gwbo[(size_t)n * 1024 + c];
        gbm[s * 1024 + c] = acc / fmaxf(cnt[8 + s], 1e-6f);
    }
}

// ---------------- gout = einsum(op, gq) + gb, as bf16 for the final GEMM --------
__global__ void gout_kernel(const float* __restrict__ cmean, const float* __restrict__ bmean,
                            const float* __restrict__ gq, const float* __restrict__ gbm,
                            const int* __restrict__ chain, const int* __restrict__ batch,
                            __bf16* __restrict__ outb)
{
    int n = blockIdx.x;
    int t = threadIdx.x;
    int cs = chain[n], bs = batch[n];
    for (int r = 0; r < 4; ++r) {
        int c = t + r * 256;              // 0..1023
        int h = c >> 7;
        int vv = c & 127;
        const float* op = (vv < 64)
            ? (cmean + (((size_t)cs * 8 + h) * 64 + vv) * 64)
            : (bmean + (((size_t)bs * 8 + h) * 64 + (vv - 64)) * 64);
        const float* qv = gq + (size_t)n * 512 + h * 64;
        float acc = 0.0f;
        for (int k = 0; k < 64; ++k) acc += op[k] * qv[k];
        acc += gbm[bs * 1024 + c];
        outb[(size_t)n * 1024 + c] = (__bf16)acc;
    }
}

// ===========================================================================
extern "C" void kernel_launch(void* const* d_in, const int* in_sizes, int n_in,
                              void* d_out, int out_size, void* d_ws, size_t ws_size,
                              hipStream_t stream)
{
    (void)in_sizes; (void)n_in; (void)out_size; (void)ws_size;
    const float* features = (const float*)d_in[0];
    const float* pos      = (const float*)d_in[1];
    const float* logits   = (const float*)d_in[2];
    const float* rln_g = (const float*)d_in[3];
    const float* rln_b = (const float*)d_in[4];
    const float* w_rel = (const float*)d_in[5];
    const float* w_i   = (const float*)d_in[6];
    const float* w_j   = (const float*)d_in[7];
    const float* pln_g = (const float*)d_in[8];
    const float* pln_b = (const float*)d_in[9];
    const float* mlp_w1 = (const float*)d_in[10];
    const float* mlp_b1 = (const float*)d_in[11];
    const float* mlp_w2 = (const float*)d_in[12];
    const float* mlp_b2 = (const float*)d_in[13];
    const float* aln_g = (const float*)d_in[14];
    const float* aln_b = (const float*)d_in[15];
    const float* wq = (const float*)d_in[16];
    const float* wk = (const float*)d_in[17];
    const float* wv = (const float*)d_in[18];
    const float* w_bias = (const float*)d_in[19];
    const float* w_o = (const float*)d_in[20];
    const float* w_rel2 = (const float*)d_in[21];
    const float* w_log  = (const float*)d_in[22];
    const float* w_rbf  = (const float*)d_in[23];
    const float* w_dir  = (const float*)d_in[24];
    const float* w_rot  = (const float*)d_in[25];
    const float* w_vec  = (const float*)d_in[26];
    const float* ppln_g = (const float*)d_in[27];
    const float* ppln_b = (const float*)d_in[28];
    const float* pmlp_w1 = (const float*)d_in[29];
    const float* pmlp_b1 = (const float*)d_in[30];
    const float* pmlp_w2 = (const float*)d_in[31];
    const float* pmlp_b2 = (const float*)d_in[32];
    const float* sln_g = (const float*)d_in[33];
    const float* sln_b = (const float*)d_in[34];
    const float* swq = (const float*)d_in[35];
    const float* swk = (const float*)d_in[36];
    const float* swv = (const float*)d_in[37];
    const float* sw_bias = (const float*)d_in[38];
    const float* sw_pv   = (const float*)d_in[39];
    const float* sw_o    = (const float*)d_in[40];
    const float* gwk = (const float*)d_in[41];
    const float* gwv = (const float*)d_in[42];
    const float* gwq = (const float*)d_in[43];
    const float* gwb = (const float*)d_in[44];
    const float* g_out_w = (const float*)d_in[45];
    const int* nb    = (const int*)d_in[46];
    const int* resi  = (const int*)d_in[47];
    const int* chain = (const int*)d_in[48];
    const int* batch = (const int*)d_in[49];
    const unsigned char* mask    = (const unsigned char*)d_in[50];
    const unsigned char* posmask = (const unsigned char*)d_in[51];
    float* out = (float*)d_out;

    // ---- carve workspace (256B-aligned slabs) ----
    char* wsb = (char*)d_ws;
    size_t off = 0;
    auto alloc = [&](size_t bytes) -> void* {
        void* p = wsb + off;
        off += (bytes + 255) & ~(size_t)255;
        return p;
    };
    // bf16 weights
    __bf16* b_wi  = (__bf16*)alloc((size_t)DIM*PD*2);
    __bf16* b_wj  = (__bf16*)alloc((size_t)DIM*PD*2);
    __bf16* b_wq  = (__bf16*)alloc((size_t)DIM*DIM*2);
    __bf16* b_wk  = (__bf16*)alloc((size_t)DIM*DIM*2);
    __bf16* b_wv  = (__bf16*)alloc((size_t)DIM*DIM*2);
    __bf16* b_wo  = (__bf16*)alloc((size_t)DIM*DIM*2);
    __bf16* b_m1  = (__bf16*)alloc((size_t)PD*2*PD*2);
    __bf16* b_m2  = (__bf16*)alloc((size_t)2*PD*PD*2);
    __bf16* b_pm1 = (__bf16*)alloc((size_t)PD*2*PD*2);
    __bf16* b_pm2 = (__bf16*)alloc((size_t)2*PD*PD*2);
    __bf16* b_swq = (__bf16*)alloc((size_t)DIM*DIM*2);
    __bf16* b_swk = (__bf16*)alloc((size_t)DIM*DIM*2);
    __bf16* b_swv = (__bf16*)alloc((size_t)DIM*DIM*2);
    __bf16* b_swpv= (__bf16*)alloc((size_t)PD*DIM*2);
    __bf16* b_swo = (__bf16*)alloc((size_t)DIM*DIM*2);
    __bf16* b_gwk = (__bf16*)alloc((size_t)DIM*512*2);
    __bf16* b_gwv = (__bf16*)alloc((size_t)DIM*512*2);
    __bf16* b_gwq = (__bf16*)alloc((size_t)DIM*512*2);
    __bf16* b_gwb = (__bf16*)alloc((size_t)DIM*1024*2);
    __bf16* b_gw  = (__bf16*)alloc((size_t)1024*DIM*2);
    // activations
    __bf16* f_bf  = (__bf16*)alloc((size_t)NRES*DIM*2);
    __bf16* x_bf  = (__bf16*)alloc((size_t)NRES*DIM*2);
    __bf16* xs_bf = (__bf16*)alloc((size_t)NRES*DIM*2);
    __bf16* f2_bf = (__bf16*)alloc((size_t)NRES*DIM*2);
    __bf16* o_bf  = (__bf16*)alloc((size_t)NRES*DIM*2);
    float*  fi    = (float*) alloc((size_t)NRES*PD*4);
    float*  fj    = (float*) alloc((size_t)NRES*PD*4);
    float*  qb    = (float*) alloc((size_t)NRES*DIM*4);
    float*  kb    = (float*) alloc((size_t)NRES*DIM*4);
    float*  vb    = (float*) alloc((size_t)NRES*DIM*4);
    float*  feat1 = (float*) alloc((size_t)NRES*DIM*4);
    float*  pairf = (float*) alloc((size_t)NK*PD*4);      // pair / ppair pre-LN (reused)
    __bf16* pln_bf= (__bf16*)alloc((size_t)NK*PD*2);      // LN-out, then MLP-out (reused 4x)
    __bf16* hid_bf= (__bf16*)alloc((size_t)NK*2*PD*2);    // MLP hidden, then pvv (reused)
    float*  Rm    = (float*) alloc((size_t)NRES*9*4);
    float*  cab   = (float*) alloc((size_t)NRES*3*4);
    float*  gkb   = (float*) alloc((size_t)NRES*512*4);
    float*  gvb   = (float*) alloc((size_t)NRES*512*4);
    float*  gqb   = (float*) alloc((size_t)NRES*512*4);
    float*  gwbo  = (float*) alloc((size_t)NRES*1024*4);
    float*  cmean = (float*) alloc((size_t)8*8*64*64*4);
    float*  bmean = (float*) alloc((size_t)4*8*64*64*4);
    float*  gbm   = (float*) alloc((size_t)4*1024*4);
    float*  cnt   = (float*) alloc((size_t)16*4);
    __bf16* goutb = (__bf16*)alloc((size_t)NRES*1024*2);

    auto cvt = [&](const float* s, __bf16* d, int n) {
        f2bf_kernel<<<(n + 255) / 256, 256, 0, stream>>>(s, d, n);
    };
    auto gemm = [&](const __bf16* A, const __bf16* B, float* oF, __bf16* oB,
                    const float* bias, const float* res, const unsigned char* rmask, int rmdiv,
                    int M, int Nn, int Kk, int gelu) {
        dim3 g(Nn / 64, M / 64);
        wmma_gemm<<<g, 128, 0, stream>>>(A, B, oF, oB, bias, res, rmask, rmdiv, M, Nn, Kk, gelu);
    };

    // ---- weight conversion ----
    cvt(w_i, b_wi, DIM*PD);      cvt(w_j, b_wj, DIM*PD);
    cvt(wq, b_wq, DIM*DIM);      cvt(wk, b_wk, DIM*DIM);   cvt(wv, b_wv, DIM*DIM);
    cvt(w_o, b_wo, DIM*DIM);
    cvt(mlp_w1, b_m1, PD*2*PD);  cvt(mlp_w2, b_m2, 2*PD*PD);
    cvt(pmlp_w1, b_pm1, PD*2*PD); cvt(pmlp_w2, b_pm2, 2*PD*PD);
    cvt(swq, b_swq, DIM*DIM);    cvt(swk, b_swk, DIM*DIM); cvt(swv, b_swv, DIM*DIM);
    cvt(sw_pv, b_swpv, PD*DIM);  cvt(sw_o, b_swo, DIM*DIM);
    cvt(gwk, b_gwk, DIM*512);    cvt(gwv, b_gwv, DIM*512); cvt(gwq, b_gwq, DIM*512);
    cvt(gwb, b_gwb, DIM*1024);   cvt(g_out_w, b_gw, 1024*DIM);

    // ---- block 1: pair features + first attention ----
    ln_kernel<<<NRES, DIM, DIM*4, stream>>>(features, rln_g, rln_b, f_bf, DIM);
    ln_kernel<<<NRES, DIM, DIM*4, stream>>>(features, aln_g, aln_b, x_bf, DIM);
    gemm(f_bf, b_wi, fi, nullptr, nullptr, nullptr, nullptr, 1, NRES, PD, DIM, 0);
    gemm(f_bf, b_wj, fj, nullptr, nullptr, nullptr, nullptr, 1, NRES, PD, DIM, 0);
    pair_build<<<NK, PD, 0, stream>>>(nb, resi, chain, batch, w_rel, fi, fj, pairf);
    ln_kernel<<<NK, PD, PD*4, stream>>>(pairf, pln_g, pln_b, pln_bf, PD);
    gemm(pln_bf, b_m1, nullptr, hid_bf, mlp_b1, nullptr, nullptr, 1, NK, 2*PD, PD, 1);
    gemm(hid_bf, b_m2, nullptr, pln_bf, mlp_b2, nullptr, nullptr, 1, NK, PD, 2*PD, 0);
    gemm(x_bf, b_wq, qb, nullptr, nullptr, nullptr, nullptr, 1, NRES, DIM, DIM, 0);
    gemm(x_bf, b_wk, kb, nullptr, nullptr, nullptr, nullptr, 1, NRES, DIM, DIM, 0);
    gemm(x_bf, b_wv, vb, nullptr, nullptr, nullptr, nullptr, 1, NRES, DIM, DIM, 0);
    attn_kernel<<<NRES, 256, 0, stream>>>(qb, kb, vb, nb, pln_bf, w_bias, nullptr,
                                          mask, nullptr, o_bf);
    gemm(o_bf, b_wo, feat1, nullptr, nullptr, features, nullptr, 1, NRES, DIM, DIM, 0);

    // ---- block 2: geometric pair features + second attention ----
    frames_kernel<<<(NRES + 63) / 64, 64, 0, stream>>>(pos, Rm, cab);
    ppair_build<<<NK, PD, 0, stream>>>(pos, logits, Rm, cab, nb, resi, chain, batch,
                                       mask, posmask, w_rel2, w_log, w_rbf, w_dir,
                                       w_rot, w_vec, pairf);
    ln_kernel<<<NK, PD, PD*4, stream>>>(pairf, ppln_g, ppln_b, pln_bf, PD);
    gemm(pln_bf, b_pm1, nullptr, hid_bf, pmlp_b1, nullptr, nullptr, 1, NK, 2*PD, PD, 1);
    gemm(hid_bf, b_pm2, nullptr, pln_bf, pmlp_b2, nullptr, posmask, KNB, NK, PD, 2*PD, 0);
    gemm(pln_bf, b_swpv, nullptr, hid_bf, nullptr, nullptr, nullptr, 1, NK, DIM, PD, 0); // pvv
    ln_kernel<<<NRES, DIM, DIM*4, stream>>>(feat1, sln_g, sln_b, xs_bf, DIM);
    gemm(xs_bf, b_swq, qb, nullptr, nullptr, nullptr, nullptr, 1, NRES, DIM, DIM, 0);
    gemm(xs_bf, b_swk, kb, nullptr, nullptr, nullptr, nullptr, 1, NRES, DIM, DIM, 0);
    gemm(xs_bf, b_swv, vb, nullptr, nullptr, nullptr, nullptr, 1, NRES, DIM, DIM, 0);
    attn_kernel<<<NRES, 256, 0, stream>>>(qb, kb, vb, nb, pln_bf, sw_bias, hid_bf,
                                          mask, posmask, o_bf);
    gemm(o_bf, b_swo, out, nullptr, nullptr, feat1, posmask, 1, NRES, DIM, DIM, 0);

    // ---- block 3: gated global update ----
    cvt(out, f2_bf, NRES*DIM);
    gemm(f2_bf, b_gwk, gkb, nullptr, nullptr, nullptr, nullptr, 1, NRES, 512, DIM, 1);
    gemm(f2_bf, b_gwv, gvb, nullptr, nullptr, nullptr, nullptr, 1, NRES, 512, DIM, 0);
    gemm(f2_bf, b_gwq, gqb, nullptr, nullptr, nullptr, nullptr, 1, NRES, 512, DIM, 1);
    gemm(f2_bf, b_gwb, gwbo, nullptr, nullptr, nullptr, nullptr, 1, NRES, 1024, DIM, 0);
    counts_kernel<<<1, 32, 0, stream>>>(chain, batch, mask, cnt);
    kvseg_kernel<<<8*8*64, 64, 0, stream>>>(gkb, gvb, chain, mask, cnt, 0, cmean);
    kvseg_kernel<<<4*8*64, 64, 0, stream>>>(gkb, gvb, batch, mask, cnt, 8, bmean);
    gbmean_kernel<<<4, 256, 0, stream>>>(gwbo, batch, mask, cnt, gbm);
    gout_kernel<<<NRES, 256, 0, stream>>>(cmean, bmean, gqb, gbm, chain, batch, goutb);
    gemm(goutb, b_gw, out, nullptr, nullptr, out, nullptr, 1, NRES, DIM, 1024, 0);
}
